// MultiheadSelfAttention_7490422964756
// MI455X (gfx1250) — matrix operand order repaired and checked by reading, not compiled, not run on hardware
//
#include <hip/hip_runtime.h>
#include <hip/hip_bf16.h>

#define EMBED 1024
#define SEQ   2048
#define NB    4
#define NH    16
#define HD    64
#define M_TOT (NB * SEQ)   // 8192

typedef __attribute__((ext_vector_type(16))) _Float16 v16h;
typedef __attribute__((ext_vector_type(8)))  float    v8f;
typedef __attribute__((ext_vector_type(8)))  _Float16 h8;
typedef __attribute__((ext_vector_type(4)))  float    f4;
typedef __attribute__((ext_vector_type(4)))  unsigned int u32x4;
typedef __attribute__((ext_vector_type(8)))  int      i32x8;
typedef __attribute__((ext_vector_type(4)))  int      i32x4;

// ---------------------------------------------------------------------------
// WMMA fragment layouts (CDNA5 16x16x32 f16, wave32):
//  A (16x32): lane<16 -> row m0+lane, one contiguous 16-elem K run k..k+15;
//             lane>=16 -> same rows, K run shifted +16.
//  B (32x16): lane%16 -> column, one contiguous 16-elem K run; lanes>=16 +16.
//  C (16x16 f32): VGPR r -> row r (+8 for lanes>=16), lane%16 -> column.
// ---------------------------------------------------------------------------

static __device__ inline v8f wmma_f32_16x16x32(v16h a, v16h b, v8f c) {
  return __builtin_amdgcn_wmma_f32_16x16x32_f16(false, a, false, b, (short)0, c,
                                                false, false);
}

static __device__ inline v16h frag_a_f16(const _Float16* __restrict__ base, int ld,
                                         int m0, int kbase, int lane) {
  const int row = m0 + (lane & 15);
  const int k0  = kbase + ((lane >> 4) << 4);
  const _Float16* p = base + (long)row * ld + k0;
  h8 lo = *(const h8*)(p);
  h8 hi = *(const h8*)(p + 8);
  v16h a;
#pragma unroll
  for (int e = 0; e < 8; ++e) { a[e] = lo[e]; a[8 + e] = hi[e]; }
  return a;
}

// Column n of B = contiguous K-run in row-major [N, ld] source (B^T layout).
static __device__ inline v16h frag_b_f16(const _Float16* __restrict__ base, int ld,
                                         int n0, int kbase, int lane) {
  const int n  = n0 + (lane & 15);
  const int k0 = kbase + ((lane >> 4) << 4);
  const _Float16* p = base + (long)n * ld + k0;
  h8 lo = *(const h8*)(p);
  h8 hi = *(const h8*)(p + 8);
  v16h b;
#pragma unroll
  for (int e = 0; e < 8; ++e) { b[e] = lo[e]; b[8 + e] = hi[e]; }
  return b;
}

// ---------------------------------------------------------------------------
// Stage 0: one-time fp32 -> f16 conversion (x, qkv_w, proj_w). 8 elems/thread.
// ---------------------------------------------------------------------------
__global__ __launch_bounds__(256) void cvt_f32_to_f16_kernel(
    const float* __restrict__ src, _Float16* __restrict__ dst) {
  const long i = ((long)blockIdx.x * blockDim.x + threadIdx.x) * 8;
  f4 x0 = *(const f4*)(src + i);
  f4 x1 = *(const f4*)(src + i + 4);
  h8 h;
#pragma unroll
  for (int e = 0; e < 4; ++e) { h[e] = (_Float16)x0[e]; h[4 + e] = (_Float16)x1[e]; }
  *(h8*)(dst + i) = h;
}

// ---------------------------------------------------------------------------
// Stage 1: QKV GEMM, f16 in/out. Block = 128 rows x 64 cols (8 waves split M,
// 4 accumulators each). q/k/v select + head index are uniform per blockIdx.y
// -> scalar branches. Q scaled by 1/sqrt(hd); V stored transposed [BH,64,S].
// ---------------------------------------------------------------------------
__global__ __launch_bounds__(256) void mha_qkv_kernel(
    const _Float16* __restrict__ xh, const _Float16* __restrict__ wh,
    const float* __restrict__ bias, _Float16* __restrict__ qbuf,
    _Float16* __restrict__ kbuf, _Float16* __restrict__ vtbuf) {
  const int lane = threadIdx.x & 31;
  const int wave = threadIdx.x >> 5;
  const int tm  = blockIdx.x * 128 + wave * 16;
  const int tn0 = blockIdx.y * 64;

  v8f acc0 = {}, acc1 = {}, acc2 = {}, acc3 = {};
  for (int k = 0; k < EMBED; k += 32) {
    v16h a = frag_a_f16(xh, EMBED, tm, k, lane);
    acc0 = wmma_f32_16x16x32(a, frag_b_f16(wh, EMBED, tn0, k, lane), acc0);
    acc1 = wmma_f32_16x16x32(a, frag_b_f16(wh, EMBED, tn0 + 16, k, lane), acc1);
    acc2 = wmma_f32_16x16x32(a, frag_b_f16(wh, EMBED, tn0 + 32, k, lane), acc2);
    acc3 = wmma_f32_16x16x32(a, frag_b_f16(wh, EMBED, tn0 + 48, k, lane), acc3);
  }
  v8f acc[4] = {acc0, acc1, acc2, acc3};

  const int pcol  = lane & 15;
  const int rbase = (lane >> 4) << 3;
  const int which = tn0 >> 10;          // uniform: 0=q, 1=k, 2=v
  const int h     = (tn0 & 1023) >> 6;  // uniform head (tn0 is 64-aligned)

#pragma unroll
  for (int nc = 0; nc < 4; ++nc) {
    const float bv = bias[tn0 + nc * 16 + pcol];
    const int e = nc * 16 + pcol;       // hd index 0..63
#pragma unroll
    for (int r = 0; r < 8; ++r) {
      const int row = tm + r + rbase;
      const int bb  = row >> 11;
      const int t   = row & (SEQ - 1);
      const int bh  = bb * NH + h;
      const float val = acc[nc][r] + bv;
      if (which == 0)
        qbuf[((long)bh * SEQ + t) * HD + e] = (_Float16)(val * 0.125f);
      else if (which == 1)
        kbuf[((long)bh * SEQ + t) * HD + e] = (_Float16)val;
      else
        vtbuf[((long)bh * HD + e) * SEQ + t] = (_Float16)val;
    }
  }
}

// ---------------------------------------------------------------------------
// Stage 2: flash attention. One wave per 16-row Q tile; 32 keys/iter:
// 4 WMMAs for S = Q K^T, online softmax (shfl_xor row reductions matching the
// C layout), P re-laid out C->A through wave-private LDS, 4 WMMAs for O += PV.
// ---------------------------------------------------------------------------
__global__ __launch_bounds__(256) void mha_attn_kernel(
    const _Float16* __restrict__ qbuf, const _Float16* __restrict__ kbuf,
    const _Float16* __restrict__ vtbuf, _Float16* __restrict__ obuf) {
  __shared__ _Float16 pshare[8][16 * 32];

  const int lane = threadIdx.x & 31;
  const int wave = threadIdx.x >> 5;
  const int bh = blockIdx.x >> 4;
  const int q0 = (blockIdx.x & 15) * 128 + wave * 16;

  const _Float16* Q  = qbuf + (long)bh * SEQ * HD;
  const _Float16* K  = kbuf + (long)bh * SEQ * HD;
  const _Float16* Vt = vtbuf + (long)bh * HD * SEQ;
  _Float16* plds = pshare[wave];

  v16h qa0 = frag_a_f16(Q, HD, q0, 0, lane);
  v16h qa1 = frag_a_f16(Q, HD, q0, 32, lane);

  float mrow[8], lrow[8];
  v8f o0 = {}, o1 = {}, o2 = {}, o3 = {};
#pragma unroll
  for (int r = 0; r < 8; ++r) { mrow[r] = -1e30f; lrow[r] = 0.0f; }

  const int pcol  = lane & 15;
  const int rbase = (lane >> 4) << 3;

  for (int j = 0; j < SEQ; j += 32) {
    v8f s0 = {}, s1 = {};
    s0 = wmma_f32_16x16x32(qa0, frag_b_f16(K, HD, j, 0, lane), s0);
    s0 = wmma_f32_16x16x32(qa1, frag_b_f16(K, HD, j, 32, lane), s0);
    s1 = wmma_f32_16x16x32(qa0, frag_b_f16(K, HD, j + 16, 0, lane), s1);
    s1 = wmma_f32_16x16x32(qa1, frag_b_f16(K, HD, j + 16, 32, lane), s1);

#pragma unroll
    for (int r = 0; r < 8; ++r) {
      float mx = fmaxf(s0[r], s1[r]);
#pragma unroll
      for (int msk = 1; msk < 16; msk <<= 1)
        mx = fmaxf(mx, __shfl_xor(mx, msk, 32));
      const float mnew = fmaxf(mrow[r], mx);
      const float p0 = __expf(s0[r] - mnew);
      const float p1 = __expf(s1[r] - mnew);
      float rs = p0 + p1;
#pragma unroll
      for (int msk = 1; msk < 16; msk <<= 1) rs += __shfl_xor(rs, msk, 32);
      const float alpha = __expf(mrow[r] - mnew);
      lrow[r] = lrow[r] * alpha + rs;
      mrow[r] = mnew;
      o0[r] *= alpha; o1[r] *= alpha; o2[r] *= alpha; o3[r] *= alpha;
      const int prow = r + rbase;
      plds[prow * 32 + pcol]      = (_Float16)p0;
      plds[prow * 32 + 16 + pcol] = (_Float16)p1;
    }

    asm volatile("s_wait_dscnt 0" ::: "memory");

    v16h pa;
    {
      const _Float16* pp = plds + (lane & 15) * 32 + ((lane >> 4) << 4);
      h8 lo = *(const h8*)(pp);
      h8 hi = *(const h8*)(pp + 8);
#pragma unroll
      for (int e = 0; e < 8; ++e) { pa[e] = lo[e]; pa[8 + e] = hi[e]; }
    }
    asm volatile("s_wait_dscnt 0" ::: "memory");

    o0 = wmma_f32_16x16x32(pa, frag_b_f16(Vt, SEQ, 0, j, lane), o0);
    o1 = wmma_f32_16x16x32(pa, frag_b_f16(Vt, SEQ, 16, j, lane), o1);
    o2 = wmma_f32_16x16x32(pa, frag_b_f16(Vt, SEQ, 32, j, lane), o2);
    o3 = wmma_f32_16x16x32(pa, frag_b_f16(Vt, SEQ, 48, j, lane), o3);
  }

  const int b_ = bh >> 4;
  const int h  = bh & 15;
#pragma unroll
  for (int r = 0; r < 8; ++r) {
    const float inv = 1.0f / lrow[r];
    const int t = q0 + r + rbase;
    _Float16* dst = obuf + ((long)(b_ * SEQ + t)) * EMBED + h * HD;
    dst[0 * 16 + pcol] = (_Float16)(o0[r] * inv);
    dst[1 * 16 + pcol] = (_Float16)(o1[r] * inv);
    dst[2 * 16 + pcol] = (_Float16)(o2[r] * inv);
    dst[3 * 16 + pcol] = (_Float16)(o3[r] * inv);
  }
}

// ---------------------------------------------------------------------------
// Stage 3: output projection with TDM-staged B tile.
// Block = 128 rows x 64 cols. Per K-step, wave 0 issues a Tensor Data Mover
// load of the 64x32 f16 weight tile into LDS (2D D#, data_size=2B), waits on
// TENSORcnt, then all 8 waves consume it. Output fp32 + bias.
// ---------------------------------------------------------------------------
__global__ __launch_bounds__(256) void mha_proj_kernel(
    const _Float16* __restrict__ oh, const _Float16* __restrict__ wh,
    const float* __restrict__ bias, float* __restrict__ out) {
  __shared__ _Float16 bsh[64 * 32];  // 4 KB B tile

  const int lane = threadIdx.x & 31;
  const int wave = threadIdx.x >> 5;
  const int tm  = blockIdx.x * 128 + wave * 16;
  const int tn0 = blockIdx.y * 64;

  // --- TDM descriptor (constant parts). D# group1 per ISA 8.4:
  //   data_size=1 (2 bytes); tensor_dim0=1024 (K), tensor_dim1=1024 (N rows);
  //   tile_dim0=32 (K), tile_dim1=64 (N rows); tensor_dim0_stride=1024.
  i32x8 g1;
  g1[0] = 1 << 16;                                   // data_size = 2B
  g1[1] = (int)((EMBED & 0xFFFF) << 16);             // tensor_dim0 lo16 @bit48
  g1[2] = (int)(((unsigned)EMBED >> 16) | ((EMBED & 0xFFFF) << 16)); // d0 hi|d1 lo
  g1[3] = (int)(((unsigned)EMBED >> 16) | (32u << 16));              // d1 hi|tile0
  g1[4] = 64;                                        // tile_dim1=64, tile_dim2=0
  g1[5] = EMBED;                                     // dim0 stride lo32
  g1[6] = 0;
  g1[7] = 0;
  const i32x4 gz4 = {0, 0, 0, 0};
  const i32x8 gz8 = {0, 0, 0, 0, 0, 0, 0, 0};
  const unsigned int ldsoff =
      (unsigned int)(size_t)(&bsh[0]);               // low 32 bits = LDS offset
  const unsigned long long gbase =
      (unsigned long long)(size_t)(wh + (long)tn0 * EMBED);

  v8f acc0 = {}, acc1 = {}, acc2 = {}, acc3 = {};
  for (int k = 0; k < EMBED; k += 32) {
    if (wave == 0) {
      const unsigned long long ga = gbase + (unsigned long long)k * 2;
      u32x4 g0;
      g0[0] = 1u;                                    // count=1, user mode
      g0[1] = ldsoff;                                // lds_addr
      g0[2] = (unsigned int)ga;                      // global_addr[31:0]
      g0[3] = (unsigned int)((ga >> 32) & 0x1FFFFFFu) | (2u << 30); // type=2
      __builtin_amdgcn_tensor_load_to_lds(g0, g1, gz4, gz4, gz8, 0);
      __builtin_amdgcn_s_wait_tensorcnt(0);
    }
    __syncthreads();

    v16h a = frag_a_f16(oh, EMBED, tm, k, lane);
    {
      const _Float16* bp0 = bsh + (0 * 16 + (lane & 15)) * 32 + ((lane >> 4) << 4);
      const _Float16* bp1 = bsh + (1 * 16 + (lane & 15)) * 32 + ((lane >> 4) << 4);
      const _Float16* bp2 = bsh + (2 * 16 + (lane & 15)) * 32 + ((lane >> 4) << 4);
      const _Float16* bp3 = bsh + (3 * 16 + (lane & 15)) * 32 + ((lane >> 4) << 4);
      v16h b;
      h8 lo, hi;
      lo = *(const h8*)bp0; hi = *(const h8*)(bp0 + 8);
#pragma unroll
      for (int e = 0; e < 8; ++e) { b[e] = lo[e]; b[8 + e] = hi[e]; }
      acc0 = wmma_f32_16x16x32(a, b, acc0);
      lo = *(const h8*)bp1; hi = *(const h8*)(bp1 + 8);
#pragma unroll
      for (int e = 0; e < 8; ++e) { b[e] = lo[e]; b[8 + e] = hi[e]; }
      acc1 = wmma_f32_16x16x32(a, b, acc1);
      lo = *(const h8*)bp2; hi = *(const h8*)(bp2 + 8);
#pragma unroll
      for (int e = 0; e < 8; ++e) { b[e] = lo[e]; b[8 + e] = hi[e]; }
      acc2 = wmma_f32_16x16x32(a, b, acc2);
      lo = *(const h8*)bp3; hi = *(const h8*)(bp3 + 8);
#pragma unroll
      for (int e = 0; e < 8; ++e) { b[e] = lo[e]; b[8 + e] = hi[e]; }
      acc3 = wmma_f32_16x16x32(a, b, acc3);
    }
    __syncthreads();
  }

  v8f acc[4] = {acc0, acc1, acc2, acc3};
  const int rbase = (lane >> 4) << 3;
#pragma unroll
  for (int nc = 0; nc < 4; ++nc) {
    const int col  = tn0 + nc * 16 + (lane & 15);
    const float bv = bias[col];
#pragma unroll
    for (int r = 0; r < 8; ++r) {
      const int row = tm + r + rbase;
      out[(long)row * EMBED + col] = acc[nc][r] + bv;
    }
  }
}

// ---------------------------------------------------------------------------
extern "C" void kernel_launch(void* const* d_in, const int* in_sizes, int n_in,
                              void* d_out, int out_size, void* d_ws,
                              size_t ws_size, hipStream_t stream) {
  const float* x      = (const float*)d_in[0];
  const float* qkv_w  = (const float*)d_in[1];
  const float* qkv_b  = (const float*)d_in[2];
  const float* proj_w = (const float*)d_in[3];
  const float* proj_b = (const float*)d_in[4];
  float* out = (float*)d_out;

  // Workspace carve-up (f16 elements):
  //   qbuf/kbuf/vtbuf/obuf: 4 x 8192*1024 (64 MB)
  //   xh: 8192*1024 (16 MB), wqkvh: 3072*1024 (6 MB), wprojh: 1024*1024 (2 MB)
  _Float16* qbuf   = (_Float16*)d_ws;
  _Float16* kbuf   = qbuf + (long)M_TOT * EMBED;
  _Float16* vtbuf  = kbuf + (long)M_TOT * EMBED;
  _Float16* obuf   = vtbuf + (long)M_TOT * EMBED;
  _Float16* xh     = obuf + (long)M_TOT * EMBED;
  _Float16* wqkvh  = xh + (long)M_TOT * EMBED;
  _Float16* wprojh = wqkvh + (long)3 * EMBED * EMBED;

  // Stage 0: one-time f16 conversion of activations and weights.
  cvt_f32_to_f16_kernel<<<(M_TOT * EMBED) / (256 * 8), 256, 0, stream>>>(x, xh);
  cvt_f32_to_f16_kernel<<<(3 * EMBED * EMBED) / (256 * 8), 256, 0, stream>>>(
      qkv_w, wqkvh);
  cvt_f32_to_f16_kernel<<<(EMBED * EMBED) / (256 * 8), 256, 0, stream>>>(
      proj_w, wprojh);

  // Stage 1: QKV GEMM, M=8192, N=3072. Blocks: 64 x 48.
  mha_qkv_kernel<<<dim3(M_TOT / 128, 3 * EMBED / 64), 256, 0, stream>>>(
      xh, wqkvh, qkv_b, qbuf, kbuf, vtbuf);

  // Stage 2: flash attention. 64 (b,h) pairs x 16 q-blocks of 128 rows.
  mha_attn_kernel<<<dim3(NB * NH * (SEQ / 128)), 256, 0, stream>>>(
      qbuf, kbuf, vtbuf, obuf);

  // Stage 3: output projection with TDM-staged weights. Blocks: 64 x 16.
  mha_proj_kernel<<<dim3(M_TOT / 128, EMBED / 64), 256, 0, stream>>>(
      obuf, wprojh, proj_b, out);
}
